// TransformerBasedPooling_77481210020099
// MI455X (gfx1250) — matrix operand order repaired and checked by reading, not compiled, not run on hardware
//
#include <hip/hip_runtime.h>

// ---------------- problem constants ----------------
#define EDIM   512
#define NHEAD  8
#define DHEAD  64
#define BN     8
#define LMAX   1920
#define LP     1936                 // Lmax+1=1921 padded to multiple of 16
#define NROWS  (BN * LP)            // 15488 padded token rows (divisible by 32)
#define QKVN   (3 * EDIM)           // 1536

typedef __attribute__((ext_vector_type(16))) _Float16 v16h;
typedef __attribute__((ext_vector_type(8)))  _Float16 v8h;
typedef __attribute__((ext_vector_type(8)))  float    v8f;

// ---------------- WMMA helpers (CDNA5 16x16x32 f16->f32) ----------------
__device__ __forceinline__ v8f wmma_ff16(v16h a, v16h b, v8f c) {
  return __builtin_amdgcn_wmma_f32_16x16x32_f16(false, a, false, b, (short)0, c,
                                                false, false);
}

// A-matrix 16x32 f16 fragment, row-major source with stride ld.
// Lanes 0-15 hold K=0..15, lanes 16-31 K=16..31 (contiguous),
// row M = lane&15.
__device__ __forceinline__ v16h load_a_frag(const _Float16* __restrict__ base,
                                            int ld, int lane) {
  const _Float16* row = base + (size_t)(lane & 15) * ld;
  const int lo = (lane & 16) ? 16 : 0;
  v16h a;
#pragma unroll
  for (int j = 0; j < 8; ++j) a[j] = row[lo + j];
#pragma unroll
  for (int j = 0; j < 8; ++j) a[8 + j] = row[8 + lo + j];
  return a;
}

// B-matrix 32x16 fragment where B[k][n] = W[n][k], W row-major NxK stride ld.
// Lane holds N = lane&15; lanes 0-15 K=0..15, lanes 16-31 K=16..31 (contiguous).
__device__ __forceinline__ v16h load_b_frag_nk(const _Float16* __restrict__ base,
                                               int ld, int lane) {
  const _Float16* row = base + (size_t)(lane & 15) * ld + ((lane & 16) ? 16 : 0);
  v16h b;
#pragma unroll
  for (int j = 0; j < 16; ++j) b[j] = row[j];
  return b;
}

// B-matrix 32x16 fragment from row-major KxN source (stride ld): B[k][n] direct.
// (Used against LDS-staged tiles, where the K-stride gather is cheap.)
__device__ __forceinline__ v16h load_b_frag_kn(const _Float16* __restrict__ base,
                                               int ld, int lane) {
  const _Float16* col = base + (lane & 15) + (size_t)((lane & 16) ? 16 : 0) * ld;
  v16h b;
#pragma unroll
  for (int j = 0; j < 16; ++j) b[j] = col[(size_t)j * ld];
  return b;
}

// ---------------- elementwise / prep kernels ----------------
__global__ void k_f32_to_f16(const float* __restrict__ src,
                             _Float16* __restrict__ dst, int n) {
  int i = blockIdx.x * blockDim.x + threadIdx.x;
  if (i < n) dst[i] = (_Float16)src[i];
}

__global__ void k_build_xp(const float* __restrict__ x,
                           const long long* __restrict__ lengths,
                           const float* __restrict__ ctok,
                           float* __restrict__ Xp) {
  size_t i = (size_t)blockIdx.x * blockDim.x + threadIdx.x;
  if (i >= (size_t)NROWS * EDIM) return;
  int e = (int)(i % EDIM);
  size_t row = i / EDIM;
  int b = (int)(row / LP), l = (int)(row % LP);
  long long off = 0; int len = 0;
  for (int q = 0; q < BN; ++q) {
    long long Lq = lengths[q];
    if (q < b) off += Lq;
    if (q == b) len = (int)Lq;
  }
  int len1 = len + 1;
  float val = 0.f;
  if (l < len1) {
    float pe = (float)l / (float)len1;
    float base = (l == 0) ? ctok[e] : x[(off + (l - 1)) * EDIM + e];
    val = base + pe;
  }
  Xp[i] = val;
}

__global__ void k_layernorm(const float* __restrict__ X,
                            const float* __restrict__ g,
                            const float* __restrict__ bta,
                            _Float16* __restrict__ Hh) {
  __shared__ float red[128];
  const int row = blockIdx.x;
  const float* x = X + (size_t)row * EDIM;
  _Float16* hh = Hh + (size_t)row * EDIM;
  const int t = threadIdx.x; // 128 threads
  float vals[4];
  float s = 0.f;
#pragma unroll
  for (int i = 0; i < 4; ++i) { vals[i] = x[t + i * 128]; s += vals[i]; }
  red[t] = s; __syncthreads();
  for (int off = 64; off > 0; off >>= 1) {
    if (t < off) red[t] += red[t + off];
    __syncthreads();
  }
  float mean = red[0] * (1.f / EDIM);
  __syncthreads();
  float vs = 0.f;
#pragma unroll
  for (int i = 0; i < 4; ++i) { float d = vals[i] - mean; vs += d * d; }
  red[t] = vs; __syncthreads();
  for (int off = 64; off > 0; off >>= 1) {
    if (t < off) red[t] += red[t + off];
    __syncthreads();
  }
  float inv = rsqrtf(red[0] * (1.f / EDIM) + 1e-5f);
#pragma unroll
  for (int i = 0; i < 4; ++i) {
    int c = t + i * 128;
    hh[c] = (_Float16)((vals[i] - mean) * inv * g[c] + bta[c]);
  }
}

// ---- WMMA tiled GEMM: C[M,N] = A[M,K](f16) * W[N,K]^T + bias -----------
// One wave computes a 32x32 tile (4 accumulators, A/B fragment reuse) with
// register double-buffering of the next k-step's fragments so global_load_b128
// latency overlaps the 4-WMMA chain.
__global__ void k_gemm_bias(const _Float16* __restrict__ A,
                            const _Float16* __restrict__ W,
                            const float* __restrict__ bias,
                            float* __restrict__ Cf, _Float16* __restrict__ Ch,
                            int M, int N, int K) {
  const int lane = threadIdx.x & 31;
  const int gw = blockIdx.x * (blockDim.x >> 5) + (threadIdx.x >> 5);
  const int tn = N >> 5, tm = M >> 5;
  if (gw >= tm * tn) return;
  const int m0 = (gw / tn) * 32, n0 = (gw % tn) * 32;
  const int cb = lane & 15;
  const float bz0 = bias ? bias[n0 + cb] : 0.f;
  const float bz1 = bias ? bias[n0 + 16 + cb] : 0.f;
  v8f c00, c01, c10, c11;
#pragma unroll
  for (int v = 0; v < 8; ++v) { c00[v] = bz0; c10[v] = bz0; c01[v] = bz1; c11[v] = bz1; }
  const _Float16* A0 = A + (size_t)m0 * K;
  const _Float16* A1 = A0 + (size_t)16 * K;
  const _Float16* W0 = W + (size_t)n0 * K;
  const _Float16* W1 = W0 + (size_t)16 * K;

  v16h a0 = load_a_frag(A0, K, lane);
  v16h a1 = load_a_frag(A1, K, lane);
  v16h b0 = load_b_frag_nk(W0, K, lane);
  v16h b1 = load_b_frag_nk(W1, K, lane);
  for (int k0 = 32; k0 < K; k0 += 32) {
    // prefetch next-k fragments before consuming current ones
    v16h na0 = load_a_frag(A0 + k0, K, lane);
    v16h na1 = load_a_frag(A1 + k0, K, lane);
    v16h nb0 = load_b_frag_nk(W0 + k0, K, lane);
    v16h nb1 = load_b_frag_nk(W1 + k0, K, lane);
    __builtin_prefetch(A0 + k0 + 256, 0, 1);  // global_prefetch_b8
    c00 = wmma_ff16(a0, b0, c00);
    c01 = wmma_ff16(a0, b1, c01);
    c10 = wmma_ff16(a1, b0, c10);
    c11 = wmma_ff16(a1, b1, c11);
    a0 = na0; a1 = na1; b0 = nb0; b1 = nb1;
  }
  c00 = wmma_ff16(a0, b0, c00);
  c01 = wmma_ff16(a0, b1, c01);
  c10 = wmma_ff16(a1, b0, c10);
  c11 = wmma_ff16(a1, b1, c11);

  const int rofs = (lane & 16) ? 8 : 0;
#pragma unroll
  for (int v = 0; v < 8; ++v) {
    size_t r0 = (size_t)(m0 + v + rofs) * N;
    size_t r1 = (size_t)(m0 + 16 + v + rofs) * N;
    if (Cf) {
      Cf[r0 + n0 + cb] = c00[v];
      Cf[r0 + n0 + 16 + cb] = c01[v];
      Cf[r1 + n0 + cb] = c10[v];
      Cf[r1 + n0 + 16 + cb] = c11[v];
    }
    if (Ch) {
      Ch[r0 + n0 + cb] = (_Float16)c00[v];
      Ch[r0 + n0 + 16 + cb] = (_Float16)c01[v];
      Ch[r1 + n0 + cb] = (_Float16)c10[v];
      Ch[r1 + n0 + 16 + cb] = (_Float16)c11[v];
    }
  }
}

// ---------------- flash attention: one wave = 16 queries of one (b,h) --------
// V tiles are staged through per-wave LDS with coalesced b128 loads; P goes
// through LDS for the C-layout -> A-layout transpose.
__global__ void k_flash(const _Float16* __restrict__ qkv,
                        const long long* __restrict__ lengths,
                        _Float16* __restrict__ Oh) {
  __shared__ _Float16 sP[8][16][34];  // per-wave 16x32 P tile (padded)
  __shared__ _Float16 sV[8][32][72];  // per-wave 32x64 V tile (padded stride)
  const int lane = threadIdx.x & 31;
  const int wv = threadIdx.x >> 5;
  const int TQ = LP / 16;
  const int gw = blockIdx.x * 8 + wv;
  if (gw >= BN * NHEAD * TQ) return;
  const int b = gw / (NHEAD * TQ);
  const int rem = gw % (NHEAD * TQ);
  const int h = rem / TQ;
  const int tq = rem % TQ;
  const int len1 = (int)lengths[b] + 1;
  const int m0 = tq * 16;
  const size_t rb = (size_t)b * LP * QKVN;
  const _Float16* qb = qkv + rb + h * DHEAD;
  const _Float16* kb = qkv + rb + EDIM + h * DHEAD;
  const _Float16* vb = qkv + rb + 2 * EDIM + h * DHEAD;
  _Float16* ob = Oh + ((size_t)b * LP + m0) * EDIM + h * DHEAD;

  if (m0 >= len1) {  // padded query tile: emit zeros so downstream GEMM is clean
    if (lane < 16) {
      _Float16* p = ob + (size_t)lane * EDIM;
      for (int j = 0; j < DHEAD; ++j) p[j] = (_Float16)0.f;
    }
    return;
  }

  const v16h qa0 = load_a_frag(qb + (size_t)m0 * QKVN, QKVN, lane);
  const v16h qa1 = load_a_frag(qb + (size_t)m0 * QKVN + 32, QKVN, lane);
  v8f o0 = {}, o1 = {}, o2 = {}, o3 = {};
  float mrow[8], lrow[8];
#pragma unroll
  for (int v = 0; v < 8; ++v) { mrow[v] = -1e30f; lrow[v] = 0.f; }
  const int colbase = lane & 15;
  const int rofs = (lane & 16) ? 8 : 0;
  const float scale = 0.125f;  // 1/sqrt(64)

  for (int j0 = 0; j0 < len1; j0 += 32) {
    const int j1 = j0 + 16;

    // stage V tile (32 keys x 64 dims) into per-wave LDS: one row per lane,
    // 8 coalesced 16B vectors; latency overlaps the S WMMAs + softmax below.
    {
      const _Float16* vrow = vb + (size_t)(j0 + lane) * QKVN;
      _Float16* drow = &sV[wv][lane][0];
#pragma unroll
      for (int d = 0; d < 8; ++d)
        *(v8h*)(drow + d * 8) = *(const v8h*)(vrow + d * 8);
    }

    // S tiles: [16q x 16k] x2, K-dim = 64 head dims
    v8f s0 = {}, s1 = {};
    s0 = wmma_ff16(qa0, load_b_frag_nk(kb + (size_t)j0 * QKVN, QKVN, lane), s0);
    s0 = wmma_ff16(qa1, load_b_frag_nk(kb + (size_t)j0 * QKVN + 32, QKVN, lane), s0);
    s1 = wmma_ff16(qa0, load_b_frag_nk(kb + (size_t)j1 * QKVN, QKVN, lane), s1);
    s1 = wmma_ff16(qa1, load_b_frag_nk(kb + (size_t)j1 * QKVN + 32, QKVN, lane), s1);

    const bool msk0 = (j0 + colbase) >= len1;
    const bool msk1 = (j1 + colbase) >= len1;
#pragma unroll
    for (int v = 0; v < 8; ++v) {
      float a0 = msk0 ? -1e9f : s0[v] * scale;
      float a1 = msk1 ? -1e9f : s1[v] * scale;
      float mx = fmaxf(a0, a1);
      mx = fmaxf(mx, __shfl_xor(mx, 1, 32));
      mx = fmaxf(mx, __shfl_xor(mx, 2, 32));
      mx = fmaxf(mx, __shfl_xor(mx, 4, 32));
      mx = fmaxf(mx, __shfl_xor(mx, 8, 32));
      float mnew = fmaxf(mrow[v], mx);
      float corr = __expf(mrow[v] - mnew);
      float p0 = __expf(a0 - mnew);
      float p1 = __expf(a1 - mnew);
      float ps = p0 + p1;
      ps += __shfl_xor(ps, 1, 32);
      ps += __shfl_xor(ps, 2, 32);
      ps += __shfl_xor(ps, 4, 32);
      ps += __shfl_xor(ps, 8, 32);
      lrow[v] = lrow[v] * corr + ps;
      mrow[v] = mnew;
      o0[v] *= corr; o1[v] *= corr; o2[v] *= corr; o3[v] *= corr;
      s0[v] = p0; s1[v] = p1;
    }
    // C-layout -> A-layout transpose of P via per-wave LDS tile
#pragma unroll
    for (int v = 0; v < 8; ++v) {
      sP[wv][v + rofs][colbase]      = (_Float16)s0[v];
      sP[wv][v + rofs][16 + colbase] = (_Float16)s1[v];
    }
    __builtin_amdgcn_wave_barrier();
    const v16h pa = load_a_frag(&sP[wv][0][0], 34, lane);
    // O += P[16x32] * V[32x64], V fragments read from LDS stage
    const _Float16* vl = &sV[wv][0][0];
    o0 = wmma_ff16(pa, load_b_frag_kn(vl + 0,  72, lane), o0);
    o1 = wmma_ff16(pa, load_b_frag_kn(vl + 16, 72, lane), o1);
    o2 = wmma_ff16(pa, load_b_frag_kn(vl + 32, 72, lane), o2);
    o3 = wmma_ff16(pa, load_b_frag_kn(vl + 48, 72, lane), o3);
    __builtin_amdgcn_wave_barrier();
  }

#pragma unroll
  for (int v = 0; v < 8; ++v) {
    float inv = 1.0f / lrow[v];
    o0[v] *= inv; o1[v] *= inv; o2[v] *= inv; o3[v] *= inv;
  }
#pragma unroll
  for (int v = 0; v < 8; ++v) {
    _Float16* orow = ob + (size_t)(v + rofs) * EDIM;
    orow[colbase]      = (_Float16)o0[v];
    orow[16 + colbase] = (_Float16)o1[v];
    orow[32 + colbase] = (_Float16)o2[v];
    orow[48 + colbase] = (_Float16)o3[v];
  }
}

// ---------------- residual / conv / output kernels ----------------
__global__ void k_residual(float* __restrict__ Xp, const float* __restrict__ Acc,
                           const long long* __restrict__ lengths, int masked) {
  size_t i = (size_t)blockIdx.x * blockDim.x + threadIdx.x;
  if (i >= (size_t)NROWS * EDIM) return;
  size_t row = i / EDIM;
  int b = (int)(row / LP), l = (int)(row % LP);
  if (masked) {
    int len1 = (int)lengths[b] + 1;
    Xp[i] = (l < len1) ? Xp[i] + Acc[i] : 0.f;
  } else {
    Xp[i] = Xp[i] + Acc[i];
  }
}

__global__ void k_conv(const float* __restrict__ Xp,
                       const long long* __restrict__ lengths,
                       const float* __restrict__ w3, const float* __restrict__ b3,
                       const float* __restrict__ w5, const float* __restrict__ b5,
                       const float* __restrict__ w7, const float* __restrict__ b7,
                       float* __restrict__ Tmp) {
  size_t i = (size_t)blockIdx.x * blockDim.x + threadIdx.x;
  if (i >= (size_t)BN * LMAX * EDIM) return;
  int c = (int)(i % EDIM);
  size_t rp = i / EDIM;
  int b = (int)(rp / LMAX), p = (int)(rp % LMAX);
  int len = (int)lengths[b];
  if (p >= len) { Tmp[i] = 0.f; return; }
  auto bv = [&](int pp) -> float {
    return (pp >= 0 && pp < len) ? Xp[((size_t)b * LP + pp + 1) * EDIM + c] : 0.f;
  };
  float y = b3[c] + b5[c] + b7[c];
#pragma unroll
  for (int t = 0; t < 3; ++t) y += w3[c * 3 + t] * bv(p + t - 1);
#pragma unroll
  for (int t = 0; t < 5; ++t) y += w5[c * 5 + t] * bv(p + t - 2);
#pragma unroll
  for (int t = 0; t < 7; ++t) y += w7[c * 7 + t] * bv(p + t - 3);
  Tmp[i] = bv(p) + y;
}

__global__ void k_conv_wb(float* __restrict__ Xp, const float* __restrict__ Tmp) {
  size_t i = (size_t)blockIdx.x * blockDim.x + threadIdx.x;
  if (i >= (size_t)BN * LMAX * EDIM) return;
  int c = (int)(i % EDIM);
  size_t rp = i / EDIM;
  int b = (int)(rp / LMAX), p = (int)(rp % LMAX);
  Xp[((size_t)b * LP + p + 1) * EDIM + c] = Tmp[i];
}

__global__ void k_outcls(const float* __restrict__ Xp, float* __restrict__ out) {
  int i = blockIdx.x * blockDim.x + threadIdx.x;
  if (i >= BN * EDIM) return;
  int b = i / EDIM, e = i % EDIM;
  out[i] = Xp[(size_t)b * LP * EDIM + e];
}

// softmax over class-token query row per (b,h) from f16 QKV of attention #2
__global__ void k_cls_probs(const _Float16* __restrict__ qkv,
                            const long long* __restrict__ lengths,
                            float* __restrict__ Z) {
  __shared__ float red[256];
  __shared__ _Float16 q0[DHEAD];
  const int bh = blockIdx.x;
  const int b = bh / NHEAD, h = bh % NHEAD;
  const int len1 = (int)lengths[b] + 1;
  const _Float16* qrow = qkv + (size_t)b * LP * QKVN + h * DHEAD;
  const _Float16* kbase = qkv + (size_t)b * LP * QKVN + EDIM + h * DHEAD;
  const int t = threadIdx.x;
  if (t < DHEAD) q0[t] = qrow[t];
  __syncthreads();
  float* zrow = Z + (size_t)bh * LP;
  float lmax = -1e30f;
  for (int j = t; j < len1; j += 256) {
    const _Float16* kr = kbase + (size_t)j * QKVN;
    float s = 0.f;
#pragma unroll
    for (int d = 0; d < DHEAD; ++d) s += (float)q0[d] * (float)kr[d];
    s *= 0.125f;
    zrow[j] = s;
    lmax = fmaxf(lmax, s);
  }
  red[t] = lmax; __syncthreads();
  for (int off = 128; off > 0; off >>= 1) {
    if (t < off) red[t] = fmaxf(red[t], red[t + off]);
    __syncthreads();
  }
  const float mx = red[0];
  __syncthreads();
  float lsum = 0.f;
  for (int j = t; j < len1; j += 256) {
    float e = __expf(zrow[j] - mx);
    zrow[j] = e;
    lsum += e;
  }
  red[t] = lsum; __syncthreads();
  for (int off = 128; off > 0; off >>= 1) {
    if (t < off) red[t] += red[t + off];
    __syncthreads();
  }
  const float inv = 1.f / red[0];
  for (int j = t; j < len1; j += 256) zrow[j] *= inv;
}

__global__ void k_aw(const float* __restrict__ Z,
                     const long long* __restrict__ lengths,
                     float* __restrict__ outaw, int total) {
  int i = blockIdx.x * blockDim.x + threadIdx.x;
  if (i >= total) return;
  long long off = 0; int b = 0;
  for (int q = 0; q < BN; ++q) {
    long long Lq = lengths[q];
    if ((long long)i < off + Lq) { b = q; break; }
    off += Lq;
  }
  int pos = i - (int)off;
  float s = 0.f;
#pragma unroll
  for (int h = 0; h < NHEAD; ++h)
    s += Z[(size_t)(b * NHEAD + h) * LP + pos + 1];
  outaw[i] = s * (1.f / NHEAD);
}

// ---------------- host orchestration ----------------
static inline int cdiv(long long a, long long b) { return (int)((a + b - 1) / b); }

extern "C" void kernel_launch(void* const* d_in, const int* in_sizes, int n_in,
                              void* d_out, int out_size, void* d_ws, size_t ws_size,
                              hipStream_t stream) {
  const float*     x       = (const float*)d_in[0];
  const long long* lengths = (const long long*)d_in[1];
  const float* ctok = (const float*)d_in[2];
  const float* ln1g = (const float*)d_in[3];
  const float* ln1b = (const float*)d_in[4];
  const float* wi1  = (const float*)d_in[5];
  const float* bi1  = (const float*)d_in[6];
  const float* wo1  = (const float*)d_in[7];
  const float* bo1  = (const float*)d_in[8];
  const float* w3   = (const float*)d_in[9];
  const float* b3   = (const float*)d_in[10];
  const float* w5   = (const float*)d_in[11];
  const float* b5   = (const float*)d_in[12];
  const float* w7   = (const float*)d_in[13];
  const float* b7   = (const float*)d_in[14];
  const float* ln2g = (const float*)d_in[15];
  const float* ln2b = (const float*)d_in[16];
  const float* wi2  = (const float*)d_in[17];
  const float* bi2  = (const float*)d_in[18];
  const float* wo2  = (const float*)d_in[19];
  const float* bo2  = (const float*)d_in[20];

  const int total = in_sizes[0] / EDIM;  // 11776 tokens
  float* out = (float*)d_out;

  // workspace carve-out (256B aligned regions)
  char* ws = (char*)d_ws;
  size_t off = 0;
  auto take = [&](size_t bytes) -> void* {
    void* p = ws + off;
    off += (bytes + 255) & ~(size_t)255;
    return p;
  };
  float*     Xp   = (float*)take((size_t)NROWS * EDIM * 4);
  _Float16*  Hh   = (_Float16*)take((size_t)NROWS * EDIM * 2);
  _Float16*  QKV  = (_Float16*)take((size_t)(NROWS + 64) * QKVN * 2);  // +64 row pad for 32-key overhang
  _Float16*  Oh   = (_Float16*)take((size_t)NROWS * EDIM * 2);
  float*     TmpF = (float*)take((size_t)NROWS * EDIM * 4);
  _Float16*  Wi1h = (_Float16*)take((size_t)QKVN * EDIM * 2);
  _Float16*  Wo1h = (_Float16*)take((size_t)EDIM * EDIM * 2);
  _Float16*  Wi2h = (_Float16*)take((size_t)QKVN * EDIM * 2);
  _Float16*  Wo2h = (_Float16*)take((size_t)EDIM * EDIM * 2);
  float*     Z    = (float*)take((size_t)BN * NHEAD * LP * 4);
  (void)ws_size; (void)n_in; (void)out_size;

  // weight conversions f32 -> f16
  k_f32_to_f16<<<cdiv(QKVN * EDIM, 256), 256, 0, stream>>>(wi1, Wi1h, QKVN * EDIM);
  k_f32_to_f16<<<cdiv(EDIM * EDIM, 256), 256, 0, stream>>>(wo1, Wo1h, EDIM * EDIM);
  k_f32_to_f16<<<cdiv(QKVN * EDIM, 256), 256, 0, stream>>>(wi2, Wi2h, QKVN * EDIM);
  k_f32_to_f16<<<cdiv(EDIM * EDIM, 256), 256, 0, stream>>>(wo2, Wo2h, EDIM * EDIM);

  const long long nelem = (long long)NROWS * EDIM;
  const int gelem = cdiv(nelem, 256);
  const int tilesQKV  = (NROWS / 32) * (QKVN / 32);   // 32x32 tiles per wave
  const int tilesProj = (NROWS / 32) * (EDIM / 32);
  const int flashWaves = BN * NHEAD * (LP / 16);
  const long long nconv = (long long)BN * LMAX * EDIM;

  // build padded input
  k_build_xp<<<gelem, 256, 0, stream>>>(x, lengths, ctok, Xp);

  // ---- block 1: LN1 -> QKV -> flash -> proj -> masked residual
  k_layernorm<<<NROWS, 128, 0, stream>>>(Xp, ln1g, ln1b, Hh);
  k_gemm_bias<<<cdiv(tilesQKV, 8), 256, 0, stream>>>(Hh, Wi1h, bi1, nullptr, QKV,
                                                     NROWS, QKVN, EDIM);
  k_flash<<<cdiv(flashWaves, 8), 256, 0, stream>>>(QKV, lengths, Oh);
  k_gemm_bias<<<cdiv(tilesProj, 8), 256, 0, stream>>>(Oh, Wo1h, bo1, TmpF, nullptr,
                                                      NROWS, EDIM, EDIM);
  k_residual<<<gelem, 256, 0, stream>>>(Xp, TmpF, lengths, 1);

  // ---- depthwise conv mixer (3,5,7) + residual on body rows
  k_conv<<<cdiv(nconv, 256), 256, 0, stream>>>(Xp, lengths, w3, b3, w5, b5, w7, b7,
                                               TmpF);
  k_conv_wb<<<cdiv(nconv, 256), 256, 0, stream>>>(Xp, TmpF);

  // ---- block 2: LN2 -> QKV -> flash -> proj -> unmasked residual
  k_layernorm<<<NROWS, 128, 0, stream>>>(Xp, ln2g, ln2b, Hh);
  k_gemm_bias<<<cdiv(tilesQKV, 8), 256, 0, stream>>>(Hh, Wi2h, bi2, nullptr, QKV,
                                                     NROWS, QKVN, EDIM);
  k_flash<<<cdiv(flashWaves, 8), 256, 0, stream>>>(QKV, lengths, Oh);
  k_gemm_bias<<<cdiv(tilesProj, 8), 256, 0, stream>>>(Oh, Wo2h, bo2, TmpF, nullptr,
                                                      NROWS, EDIM, EDIM);
  k_residual<<<gelem, 256, 0, stream>>>(Xp, TmpF, lengths, 0);

  // ---- outputs
  k_outcls<<<cdiv(BN * EDIM, 256), 256, 0, stream>>>(Xp, out);
  k_cls_probs<<<BN * NHEAD, 256, 0, stream>>>(QKV, lengths, Z);
  k_aw<<<cdiv(total, 256), 256, 0, stream>>>(Z, lengths, out + BN * EDIM, total);
}
